// SetAbstractionFast_85804856639961
// MI455X (gfx1250) — matrix operand order repaired and checked by reading, not compiled, not run on hardware
//
#include <hip/hip_runtime.h>
#include <cstdint>
#include <cstddef>

// ---------------------------------------------------------------------------
// Problem constants (from reference): B=8, N=4096, S=1024, K=20, D=29
// x = concat(xyz, points): 32 ch; feat = concat(nbr-ctr, ctr): 64 ch
// MLP: 64 -> 64 -> 64 -> 128, global BN over (B,N,K), relu, max-pool over K.
// ---------------------------------------------------------------------------
#define B_    8
#define N_    4096
#define S_    1024
#define KNN_  20
#define CPT_  32
#define ROWS_ (B_ * N_ * KNN_)          // 655360 flattened (b,n,k) rows
#define GEMM_BLOCKS (ROWS_ / 128)       // 5120 (8 waves x 16 rows per block)
#define EPS_  1e-5f

typedef __attribute__((ext_vector_type(16))) _Float16 v16h;
typedef __attribute__((ext_vector_type(8)))  _Float16 v8h;
typedef __attribute__((ext_vector_type(8)))  float    v8f;
typedef __attribute__((ext_vector_type(2)))  float    v2f;

// ---------------------------------------------------------------------------
// CDNA5 async global->LDS copy (ASYNCcnt-tracked, no VGPR data movement).
// LDS operand is the wave-relative byte offset = low 32 bits of the generic
// address (LDS aperture keeps the offset in addr[31:0] per ISA 10.2).
// ---------------------------------------------------------------------------
__device__ __forceinline__ void async_copy_b128(void* ldsDst, const void* gSrc) {
  uint32_t loff = (uint32_t)(uintptr_t)ldsDst;
  uint64_t gaddr = (uint64_t)(uintptr_t)gSrc;
  asm volatile("global_load_async_to_lds_b128 %0, %1, off"
               :: "v"(loff), "v"(gaddr) : "memory");
}
__device__ __forceinline__ void wait_async_le1() {
  asm volatile("s_wait_asynccnt 0x1" ::: "memory");
}
__device__ __forceinline__ void wait_async_le0() {
  asm volatile("s_wait_asynccnt 0x0" ::: "memory");
}

// ---------------------------------------------------------------------------
// Workspace layout (bytes)
// ---------------------------------------------------------------------------
constexpr size_t wsAlign(size_t x) { return (x + 255) & ~size_t(255); }
constexpr size_t OFF_X    = 0;
constexpr size_t SZ_X     = (size_t)B_ * N_ * CPT_ * 4;        // x fp32
constexpr size_t OFF_XX   = wsAlign(OFF_X + SZ_X);
constexpr size_t SZ_XX    = (size_t)B_ * N_ * 4;               // |x|^2
constexpr size_t OFF_IDX  = wsAlign(OFF_XX + SZ_XX);
constexpr size_t SZ_IDX   = (size_t)B_ * N_ * KNN_ * 4;        // knn idx
constexpr size_t OFF_H0   = wsAlign(OFF_IDX + SZ_IDX);
constexpr size_t SZ_H0    = (size_t)ROWS_ * 64 * 2;            // pre-BN f16
constexpr size_t OFF_H1   = wsAlign(OFF_H0 + SZ_H0);
constexpr size_t SZ_H1    = SZ_H0;
constexpr size_t OFF_H2   = wsAlign(OFF_H1 + SZ_H1);
constexpr size_t SZ_H2    = (size_t)ROWS_ * 128 * 2;
constexpr size_t OFF_PART = wsAlign(OFF_H2 + SZ_H2);
constexpr size_t SZ_PART  = (size_t)GEMM_BLOCKS * 2 * 128 * 4; // per-block sum/sumsq
constexpr size_t OFF_SS0  = wsAlign(OFF_PART + SZ_PART);       // scale/shift pairs
constexpr size_t OFF_SS1  = OFF_SS0 + 1024;
constexpr size_t OFF_SS2  = OFF_SS1 + 1024;

// ---------------------------------------------------------------------------
// Kernel 1: build x = concat(xyz, points) (fp32) and xx = sum(x*x)
// ---------------------------------------------------------------------------
__global__ __launch_bounds__(256) void build_x_kernel(
    const float* __restrict__ xyz, const float* __restrict__ pts,
    float* __restrict__ xw, float* __restrict__ xxw) {
  int t = blockIdx.x * blockDim.x + threadIdx.x;
  if (t >= B_ * N_) return;
  float acc = 0.f;
  float* dst = xw + (size_t)t * CPT_;
  const float* p3 = xyz + (size_t)t * 3;
#pragma unroll
  for (int c = 0; c < 3; ++c) { float v = p3[c]; dst[c] = v; acc += v * v; }
  const float* pd = pts + (size_t)t * 29;
#pragma unroll
  for (int c = 0; c < 29; ++c) { float v = pd[c]; dst[3 + c] = v; acc += v * v; }
  xxw[t] = acc;
}

// ---------------------------------------------------------------------------
// Kernel 2: fused Gram-matrix (f32 WMMA 16x16x4) + streaming top-20 per row.
// Block = 256 threads (8 waves) owning 128 rows of one batch. Column tiles
// of 32 points are double-buffered in LDS via async global->LDS B128 copies
// (one async instruction per wave per chunk; s_wait_asynccnt 1 overlaps the
// next chunk's HBM fetch with this chunk's WMMA + top-k work).
// ---------------------------------------------------------------------------
__global__ __launch_bounds__(256) void knn_kernel(
    const float* __restrict__ xw, const float* __restrict__ xxw,
    int* __restrict__ idxw) {
  __shared__ __align__(16) float colL[2][32 * CPT_];   // 2 x 4KB
  __shared__ __align__(16) float scoreL[8 * 16 * 16];  // 8KB

  const int tid  = threadIdx.x;
  const int wave = tid >> 5;
  const int lane = tid & 31;
  const int b    = blockIdx.x >> 5;     // / (N/128)
  const int rblk = blockIdx.x & 31;
  const int row0 = rblk * 128 + wave * 16;  // within batch

  const float* xb  = xw  + (size_t)b * N_ * CPT_;
  const float* xxb = xxw + (size_t)b * N_;

  const int mrow = lane & 15;
  const int koff = (lane < 16) ? 0 : 2;   // f32 A/B 16x16x4 lane K offset

  // Preload A fragments for this wave's 16 rows (K = 32 => 8 steps of 4).
  v2f apair[8];
  {
    const float* rp = xb + (size_t)(row0 + mrow) * CPT_;
#pragma unroll
    for (int kc = 0; kc < 8; ++kc)
      apair[kc] = *(const v2f*)(rp + kc * 4 + koff);
  }
  const float xr = xxb[row0 + mrow];

  float bs[KNN_]; int bi[KNN_];
#pragma unroll
  for (int q = 0; q < KNN_; ++q) { bs[q] = -3.0e38f; bi[q] = 0; }

  float* sw = scoreL + wave * 256;

  // Prologue: stage chunk 0 (each thread copies 16B -> 1 async op per wave).
  async_copy_b128(&colL[0][tid * 4], xb + (size_t)tid * 4);

  for (int cc = 0; cc < N_ / 32; ++cc) {
    const int cb  = cc * 32;
    const int cur = cc & 1;
    if (cc + 1 < N_ / 32) {   // stage next chunk into the other buffer
      async_copy_b128(&colL[1 - cur][tid * 4],
                      xb + (size_t)(cb + 32) * CPT_ + tid * 4);
      wait_async_le1();       // chunk cc complete (in-order), cc+1 in flight
    } else {
      wait_async_le0();
    }
    __syncthreads();

    // Per-lane column norms for this chunk (folded into score at spill time).
    const float xc0 = xxb[cb + (lane & 15)];
    const float xc1 = xxb[cb + 16 + (lane & 15)];

    for (int ct = 0; ct < 2; ++ct) {
      const float* colp = colL[cur] + (ct * 16 + mrow) * CPT_;
      v8f acc = {};
#pragma unroll
      for (int kc = 0; kc < 8; ++kc) {
        v2f bv = *(const v2f*)(colp + kc * 4 + koff);
        acc = __builtin_amdgcn_wmma_f32_16x16x4_f32(
            false, apair[kc], false, bv, (short)0, acc, false, false);
      }
      // Spill 16x16 tile with -|x_col|^2 folded in (column = lane & 15).
      const float xc = ct ? xc1 : xc0;
      if (lane < 16) {
#pragma unroll
        for (int i = 0; i < 8; ++i) sw[i * 16 + lane] = 2.f * acc[i] - xc;
      } else {
#pragma unroll
        for (int i = 0; i < 8; ++i) sw[(8 + i) * 16 + (lane - 16)] = 2.f * acc[i] - xc;
      }
      __syncthreads();
      // Lanes 0..15: one row each; streaming sorted top-20 insertion.
      if (lane < 16) {
#pragma unroll
        for (int j = 0; j < 16; ++j) {
          float s = sw[lane * 16 + j] - xr;
          int id = cb + ct * 16 + j;
          if (s > bs[KNN_ - 1]) {
            bs[KNN_ - 1] = s; bi[KNN_ - 1] = id;
#pragma unroll
            for (int q = KNN_ - 1; q > 0; --q) {
              if (bs[q] > bs[q - 1]) {
                float tf = bs[q]; bs[q] = bs[q - 1]; bs[q - 1] = tf;
                int   ti = bi[q]; bi[q] = bi[q - 1]; bi[q - 1] = ti;
              }
            }
          }
        }
      }
      __syncthreads();   // last one doubles as end-of-chunk barrier
    }
  }
  if (lane < 16) {
    int* op = idxw + ((size_t)b * N_ + row0 + mrow) * KNN_;
#pragma unroll
    for (int q = 0; q < KNN_; ++q) op[q] = bi[q];
  }
}

// ---------------------------------------------------------------------------
// Kernel 3: fused (prev-BN + relu) -> GEMM (f16 WMMA 16x16x32) -> +bias
//           -> store pre-BN f16 -> deterministic per-block channel stats.
// MODE 0: build feat = concat(nbr - ctr, ctr) from x + knn idx.
// MODE 1: act = relu(hin * scale + shift).
// Block = 256 threads = 8 waves, each wave handles 16 rows, all COUT columns.
// ---------------------------------------------------------------------------
template <int CIN, int COUT, int MODE>
__global__ __launch_bounds__(256) void gemm_kernel(
    const float* __restrict__ xw, const int* __restrict__ idxw,
    const _Float16* __restrict__ hin, const float* __restrict__ ssin,
    const float* __restrict__ Wg, const float* __restrict__ bg,
    _Float16* __restrict__ hout, float* __restrict__ part) {
  __shared__ __align__(32) _Float16 Wl[COUT * CIN];
  __shared__ float    biasL[COUT];
  __shared__ float    scL[CIN];
  __shared__ float    shL[CIN];
  __shared__ __align__(32) _Float16 actL[8 * 16 * CIN];
  __shared__ float    psumL[16 * COUT];
  __shared__ float    psqL[16 * COUT];

  const int tid = threadIdx.x, wave = tid >> 5, lane = tid & 31;

  for (int i = tid; i < COUT * CIN; i += 256) Wl[i] = (_Float16)Wg[i];
  if (tid < COUT) biasL[tid] = bg[tid];
  if constexpr (MODE == 1) {
    if (tid < CIN) { scL[tid] = ssin[tid]; shL[tid] = ssin[CIN + tid]; }
  }
  __syncthreads();

  const int r0 = (blockIdx.x * 8 + wave) * 16;

  // Build this wave's 16xCIN activation tile in LDS (one half-row per lane, 2 lanes per row).
  {
    const int m = lane >> 1, half_ = lane & 1;
    const int r = r0 + m;
    _Float16* arow = actL + (wave * 16 + m) * CIN;
    if constexpr (MODE == 0) {
      const int nn  = (r / KNN_) % N_;
      const int bb  = r / (KNN_ * N_);
      const int nbr = idxw[r];
      const float* ctrp = xw + ((size_t)bb * N_ + nn)  * CPT_;
      const float* nbrp = xw + ((size_t)bb * N_ + nbr) * CPT_;
      if (half_ == 0) {
#pragma unroll
        for (int c = 0; c < 32; ++c) arow[c] = (_Float16)(nbrp[c] - ctrp[c]);
      } else {
#pragma unroll
        for (int c = 0; c < 32; ++c) arow[32 + c] = (_Float16)ctrp[c];
      }
    } else {
      const _Float16* hp = hin + (size_t)r * CIN;
      const int cbase = half_ * (CIN / 2);
#pragma unroll
      for (int c = 0; c < CIN / 2; ++c) {
        int ch = cbase + c;
        float v = fmaf((float)hp[ch], scL[ch], shL[ch]);
        arow[ch] = (_Float16)fmaxf(v, 0.f);
      }
    }
  }
  __syncthreads();

  const int ncol = lane & 15;
  const int abase = (wave * 16 + ncol) * CIN;        // A row for this lane
  const int akoff = (lane < 16) ? 0 : 16;            // f16 A 16x32 layout
  const int bkoff = (lane < 16) ? 0 : 16;            // f16 B 32x16 layout

#pragma unroll
  for (int nt = 0; nt < COUT / 16; ++nt) {
    v8f acc = {};
#pragma unroll
    for (int kc = 0; kc < CIN / 32; ++kc) {
      // A fragment: two contiguous 16B runs (K 0-7 / 8-15 or 16-23 / 24-31).
      const _Float16* ap = actL + abase + kc * 32 + akoff;
      v8h alo = *(const v8h*)(ap);
      v8h ahi = *(const v8h*)(ap + 8);
      v16h a = __builtin_shufflevector(alo, ahi, 0, 1, 2, 3, 4, 5, 6, 7,
                                       8, 9, 10, 11, 12, 13, 14, 15);
      // B fragment: one contiguous 32B run of W row (column = output chan).
      const _Float16* bp = Wl + (nt * 16 + ncol) * CIN + kc * 32 + bkoff;
      v16h bm = *(const v16h*)(bp);
      acc = __builtin_amdgcn_wmma_f32_16x16x32_f16(
          false, a, false, bm, (short)0, acc, false, false);
    }
    const int o = nt * 16 + ncol;
    const float bv = biasL[o];
    float s = 0.f, q = 0.f;
#pragma unroll
    for (int i = 0; i < 8; ++i) { acc[i] += bv; s += acc[i]; q += acc[i] * acc[i]; }
    const int rbase = r0 + ((lane < 16) ? 0 : 8);
#pragma unroll
    for (int i = 0; i < 8; ++i)
      hout[(size_t)(rbase + i) * COUT + o] = (_Float16)acc[i];
    const int slot = wave * 2 + (lane >> 4);          // unique (slot, o) writer
    psumL[slot * COUT + o] = s;
    psqL[slot * COUT + o]  = q;
  }
  __syncthreads();

  // Deterministic fixed-order reduction of the 16 slots -> per-block partials.
  for (int o = tid; o < COUT; o += 256) {
    float s = 0.f, q = 0.f;
#pragma unroll
    for (int t2 = 0; t2 < 16; ++t2) { s += psumL[t2 * COUT + o]; q += psqL[t2 * COUT + o]; }
    part[(size_t)blockIdx.x * 2 * COUT + o]        = s;
    part[(size_t)blockIdx.x * 2 * COUT + COUT + o] = q;
  }
}

// ---------------------------------------------------------------------------
// Kernel 4: reduce block partials (fixed order) -> BN scale/shift per channel
// ---------------------------------------------------------------------------
template <int COUT>
__global__ __launch_bounds__(128) void finalize_kernel(
    const float* __restrict__ part, const float* __restrict__ g,
    const float* __restrict__ be, float* __restrict__ ss) {
  int o = threadIdx.x;
  if (o >= COUT) return;
  float s = 0.f, q = 0.f;
  for (int blk = 0; blk < GEMM_BLOCKS; ++blk) {
    s += part[(size_t)blk * 2 * COUT + o];
    q += part[(size_t)blk * 2 * COUT + COUT + o];
  }
  const float inv  = 1.0f / (float)ROWS_;
  const float mean = s * inv;
  const float var  = fmaxf(q * inv - mean * mean, 0.f);
  const float sc   = g[o] * rsqrtf(var + EPS_);
  ss[o]        = sc;
  ss[COUT + o] = be[o] - mean * sc;
}

// ---------------------------------------------------------------------------
// Kernel 5: gather sampled rows; max over K of relu(h2*scale+shift); xyz copy
// ---------------------------------------------------------------------------
__global__ __launch_bounds__(128) void output_kernel(
    const float* __restrict__ xyz, const int* __restrict__ samp,
    const _Float16* __restrict__ h2, const float* __restrict__ ss2,
    float* __restrict__ out) {
  const int bsi = blockIdx.x;            // b * S + s
  const int b = bsi / S_, s = bsi % S_;
  const int n = samp[s];
  const int o = threadIdx.x;
  float* outXyz = out;
  float* outPts = out + (size_t)B_ * S_ * 3;
  if (o < 3) outXyz[(size_t)bsi * 3 + o] = xyz[((size_t)b * N_ + n) * 3 + o];
  const _Float16* hp = h2 + ((size_t)b * N_ + n) * KNN_ * 128 + o;
  const float sc = ss2[o], sh = ss2[128 + o];
  float m = -3.0e38f;
#pragma unroll
  for (int k = 0; k < KNN_; ++k)
    m = fmaxf(m, fmaf((float)hp[(size_t)k * 128], sc, sh));
  outPts[(size_t)bsi * 128 + o] = fmaxf(m, 0.f);   // relu commutes with max
}

// ---------------------------------------------------------------------------
// Host-side launch sequence (graph-capture safe: stream-only, no allocs).
// ---------------------------------------------------------------------------
extern "C" void kernel_launch(void* const* d_in, const int* in_sizes, int n_in,
                              void* d_out, int out_size, void* d_ws, size_t ws_size,
                              hipStream_t stream) {
  (void)in_sizes; (void)n_in; (void)out_size; (void)ws_size;
  const float* xyz  = (const float*)d_in[0];
  const float* pts  = (const float*)d_in[1];
  const int*   samp = (const int*)d_in[2];
  const float* W0 = (const float*)d_in[3];
  const float* b0 = (const float*)d_in[4];
  const float* g0 = (const float*)d_in[5];
  const float* be0= (const float*)d_in[6];
  const float* W1 = (const float*)d_in[7];
  const float* b1 = (const float*)d_in[8];
  const float* g1 = (const float*)d_in[9];
  const float* be1= (const float*)d_in[10];
  const float* W2 = (const float*)d_in[11];
  const float* b2 = (const float*)d_in[12];
  const float* g2 = (const float*)d_in[13];
  const float* be2= (const float*)d_in[14];

  char* ws = (char*)d_ws;
  float*    xw   = (float*)(ws + OFF_X);
  float*    xxw  = (float*)(ws + OFF_XX);
  int*      idxw = (int*)(ws + OFF_IDX);
  _Float16* h0   = (_Float16*)(ws + OFF_H0);
  _Float16* h1   = (_Float16*)(ws + OFF_H1);
  _Float16* h2   = (_Float16*)(ws + OFF_H2);
  float*    part = (float*)(ws + OFF_PART);
  float*    ss0  = (float*)(ws + OFF_SS0);
  float*    ss1  = (float*)(ws + OFF_SS1);
  float*    ss2  = (float*)(ws + OFF_SS2);
  float*    out  = (float*)d_out;

  build_x_kernel<<<(B_ * N_ + 255) / 256, 256, 0, stream>>>(xyz, pts, xw, xxw);
  knn_kernel<<<B_ * (N_ / 128), 256, 0, stream>>>(xw, xxw, idxw);

  gemm_kernel<64, 64, 0><<<GEMM_BLOCKS, 256, 0, stream>>>(
      xw, idxw, nullptr, nullptr, W0, b0, h0, part);
  finalize_kernel<64><<<1, 128, 0, stream>>>(part, g0, be0, ss0);

  gemm_kernel<64, 64, 1><<<GEMM_BLOCKS, 256, 0, stream>>>(
      nullptr, nullptr, h0, ss0, W1, b1, h1, part);
  finalize_kernel<64><<<1, 128, 0, stream>>>(part, g1, be1, ss1);

  gemm_kernel<64, 128, 1><<<GEMM_BLOCKS, 256, 0, stream>>>(
      nullptr, nullptr, h1, ss1, W2, b2, h2, part);
  finalize_kernel<128><<<1, 128, 0, stream>>>(part, g2, be2, ss2);

  output_kernel<<<B_ * S_, 128, 0, stream>>>(xyz, samp, h2, ss2, out);
}
